// GraphLA_3994319585552
// MI455X (gfx1250) — matrix operand order repaired and checked by reading, not compiled, not run on hardware
//
#include <hip/hip_runtime.h>
#include <hip/hip_bf16.h>
#include <math.h>

// ---------------------------------------------------------------------------
// Types for CDNA5 WMMA (wave32): 16x16x32 bf16 -> f32
// ---------------------------------------------------------------------------
typedef __attribute__((ext_vector_type(16))) __bf16 v16bf;
typedef __attribute__((ext_vector_type(8)))  float  v8f;

#define WAVE 32

// ---------------------------------------------------------------------------
// Utility: fill a float buffer with a constant
// ---------------------------------------------------------------------------
__global__ void fill_kernel(float* __restrict__ p, float v, long n) {
    long i = (long)blockIdx.x * blockDim.x + threadIdx.x;
    long stride = (long)gridDim.x * blockDim.x;
    for (; i < n; i += stride) p[i] = v;
}

// ---------------------------------------------------------------------------
// Degree accumulation: deg[dst[e]] += ew[e]   (deg pre-filled with 1.0 = self loop)
// ---------------------------------------------------------------------------
__global__ void deg_kernel(const long long* __restrict__ dst,
                           const float* __restrict__ ew,
                           float* __restrict__ deg, long E) {
    long e = (long)blockIdx.x * blockDim.x + threadIdx.x;
    if (e < E) atomicAdd(&deg[dst[e]], ew[e]);
}

// deg -> dinv = rsqrt(deg)
__global__ void dinv_kernel(float* __restrict__ deg, int n) {
    int i = blockIdx.x * blockDim.x + threadIdx.x;
    if (i < n) {
        float d = deg[i];
        deg[i] = (d > 0.0f) ? rsqrtf(d) : 0.0f;
    }
}

// ---------------------------------------------------------------------------
// WMMA GEMM:  C[M,NOUT] = A[M,128] @ W[128,NOUT]  (+ optional bias[NOUT])
//
// Block = 256 threads = 8 waves. A block owns ROWS = 16*(8/TILES_N) rows of A:
//   - A stripe staged into LDS ONCE per block via GLOBAL_LOAD_ASYNC_TO_LDS_B128
//     (all TILES_N waves of a row-tile share it; previously fetched 8x from HBM)
//   - W staged into LDS as pre-packed bf16 B-fragments so each lane's 16 values
//     are 32 contiguous bytes (ds_load_b128 x2 per K-step)
// One wave computes one 16x16 tile; K=128 as 4 x WMMA(K=32).
// Fragment layouts per ISA 7.12.2 (16-bit wave32):
//   A 16x32: half h=l/16, row m=l%16; VGPR j<4 -> K=2j+8h, j>=4 -> K=16+2(j-4)+8h
//   B 32x16: half h, col n=l%16; VGPR j<4 -> K=2j+8h, j>=4 -> K=16+2(j-4)+8h
// ---------------------------------------------------------------------------
template <int NOUT>
__global__ void gemm_wmma_kernel(const float* __restrict__ A,
                                 const float* __restrict__ W,
                                 const float* __restrict__ bias,
                                 float* __restrict__ C, int M) {
    constexpr int TILES_N    = NOUT / 16;        // 8 or 4
    constexpr int MT_PER_BLK = 8 / TILES_N;      // 1 or 2
    constexpr int ROWS       = 16 * MT_PER_BLK;  // 16 or 32

    __shared__ __align__(16) float  Alds[ROWS * 128];
    __shared__ __align__(32) __bf16 Wfrag[TILES_N * 4 * 32 * 16];

    const int tid  = threadIdx.x;
    const int lane = tid & 31;
    const int wave = tid >> 5;
    const int rl   = lane & 15;
    const int half = lane >> 4;

    const int tmBase   = blockIdx.x * MT_PER_BLK;
    const int rowStart = tmBase * 16;
    const int validRows = (M - rowStart) < ROWS ? (M - rowStart) : ROWS;

    // ---- async-stage this block's A stripe (f32) into LDS ----
    {
        const int validBytes = validRows * 128 * 4;
        for (int off = tid * 16; off < validBytes; off += 256 * 16) {
            unsigned ldsAddr = (unsigned)(unsigned long long)(const void*)Alds + (unsigned)off;
            const float* g = A + (long)rowStart * 128 + (off >> 2);
            asm volatile("global_load_async_to_lds_b128 %0, %1, off"
                         :: "v"(ldsAddr), "v"(g) : "memory");
        }
    }

    // ---- stage W as bf16 B-fragments: Wfrag[((tn*4+kt)*32+lane)*16 + e] ----
    for (int t = tid; t < TILES_N * 4 * 32; t += 256) {
        const int tn = t / (4 * 32);
        const int kt = (t >> 5) & 3;
        const int ln = t & 31;
        const int c  = tn * 16 + (ln & 15);
        const int hh = ln >> 4;
        #pragma unroll
        for (int j = 0; j < 8; ++j) {
            const int k = kt * 32 + ((j < 4) ? (2 * j + 8 * hh) : (16 + 2 * (j - 4) + 8 * hh));
            Wfrag[t * 16 + 2 * j]     = (__bf16)W[k * NOUT + c];
            Wfrag[t * 16 + 2 * j + 1] = (__bf16)W[(k + 1) * NOUT + c];
        }
    }

    asm volatile("s_wait_asynccnt 0x0" ::: "memory");
    __syncthreads();

    // ---- per-wave 16x16 tile ----
    const int tnL = wave % TILES_N;              // local N tile
    const int mW  = (wave / TILES_N) * 16;       // local row base in LDS
    const int tm  = tmBase + (wave / TILES_N);
    const bool valid = (tm * 16) < M;            // partial last block (NOUT=64)
    const int col = tnL * 16 + rl;

    v8f acc = {};
    #pragma unroll
    for (int kt = 0; kt < 4; ++kt) {
        const int kbase = kt * 32;
        v16bf a;
        #pragma unroll
        for (int j = 0; j < 8; ++j) {
            const int k = kbase + ((j < 4) ? (2 * j + 8 * half)
                                           : (16 + 2 * (j - 4) + 8 * half));
            const float2 v = *(const float2*)&Alds[(mW + rl) * 128 + k];
            a[2 * j]     = (__bf16)v.x;
            a[2 * j + 1] = (__bf16)v.y;
        }
        const v16bf b = *(const v16bf*)&Wfrag[((tnL * 4 + kt) * 32 + lane) * 16];
        acc = __builtin_amdgcn_wmma_f32_16x16x32_bf16(
                  false, a, false, b, (short)0, acc, false, false);
    }

    const float bv = bias ? bias[col] : 0.0f;
    #pragma unroll
    for (int r = 0; r < 8; ++r) {
        const int orow = tm * 16 + r + 8 * half;
        if (valid && orow < M) C[(long)orow * NOUT + col] = acc[r] + bv;
    }
}

// ---------------------------------------------------------------------------
// Edge scatter: agg[dst] += dinv[src]*ew*dinv[dst] * h[src]   (wave per edge)
// ---------------------------------------------------------------------------
template <int F>
__global__ void scatter_kernel(const float* __restrict__ h,
                               float* __restrict__ agg,
                               const long long* __restrict__ src,
                               const long long* __restrict__ dst,
                               const float* __restrict__ ew,
                               const float* __restrict__ dinv,
                               long E) {
    const int lane = threadIdx.x & (WAVE - 1);
    long wave = ((long)blockIdx.x * blockDim.x + threadIdx.x) >> 5;
    const long nwaves = ((long)gridDim.x * blockDim.x) >> 5;
    constexpr int PER = F / WAVE;   // floats per lane (4 or 2)

    for (long e = wave; e < E; e += nwaves) {
        long ne = e + nwaves;
        if (ne < E) {
            long ps = src[ne];
            __builtin_prefetch(h + ps * F + lane * PER, 0, 0);  // global_prefetch_b8
        }
        const long s = src[e];
        const long d = dst[e];
        const float nrm = dinv[s] * ew[e] * dinv[d];
        if (PER == 4) {
            const float4 v = *(const float4*)(h + s * F + lane * 4);
            float* o = agg + d * F + lane * 4;
            atomicAdd(o + 0, nrm * v.x);
            atomicAdd(o + 1, nrm * v.y);
            atomicAdd(o + 2, nrm * v.z);
            atomicAdd(o + 3, nrm * v.w);
        } else {
            const float2 v = *(const float2*)(h + s * F + lane * 2);
            float* o = agg + d * F + lane * 2;
            atomicAdd(o + 0, nrm * v.x);
            atomicAdd(o + 1, nrm * v.y);
        }
    }
}

// ---------------------------------------------------------------------------
// Conv epilogue: out = relu(agg + dinv[i]^2 * h + b)   (out may alias agg)
// ---------------------------------------------------------------------------
__global__ void epilogue_kernel(const float* __restrict__ agg,
                                const float* __restrict__ h,
                                const float* __restrict__ b,
                                const float* __restrict__ dinv,
                                float* __restrict__ out,
                                long n, int F) {
    long i = (long)blockIdx.x * blockDim.x + threadIdx.x;
    if (i >= n) return;
    long node = i / F;
    int  f    = (int)(i - node * F);
    float di  = dinv[node];
    float v   = agg[i] + di * di * h[i] + b[f];
    out[i] = v > 0.0f ? v : 0.0f;
}

// ---------------------------------------------------------------------------
// Leader score: ls[i] = sigmoid( dot(ls_h[i,0:64], Wl2) + bl2 )  (wave/node)
// ---------------------------------------------------------------------------
__global__ void leader_kernel(const float* __restrict__ ls_h,
                              const float* __restrict__ Wl2,
                              const float* __restrict__ bl2,
                              float* __restrict__ ls, int n) {
    const int lane = threadIdx.x & (WAVE - 1);
    int node = (int)(((long)blockIdx.x * blockDim.x + threadIdx.x) >> 5);
    if (node >= n) return;
    const float2 hv = *(const float2*)(ls_h + (long)node * 64 + lane * 2);
    const float2 wv = *(const float2*)(Wl2 + lane * 2);
    float p = hv.x * wv.x + hv.y * wv.y;
    #pragma unroll
    for (int off = 16; off > 0; off >>= 1) p += __shfl_xor(p, off, WAVE);
    if (lane == 0) {
        float z = p + bl2[0];
        ls[node] = 1.0f / (1.0f + expf(-z));
    }
}

// ---------------------------------------------------------------------------
// Gated fusion (in place into h_nb): h_new = (1-ls)*h_nb + ls*h_self
// ---------------------------------------------------------------------------
__global__ void fusion_kernel(float* __restrict__ h_nb,
                              const float* __restrict__ h_self,
                              const float* __restrict__ ls, long n) {
    long i = (long)blockIdx.x * blockDim.x + threadIdx.x;
    if (i >= n) return;
    float g = ls[i >> 7];   // F = 128
    h_nb[i] = (1.0f - g) * h_nb[i] + g * h_self[i];
}

// ---------------------------------------------------------------------------
// Host-side orchestration
// ---------------------------------------------------------------------------
static inline int blocksFor(long n, int t) { return (int)((n + t - 1) / t); }

extern "C" void kernel_launch(void* const* d_in, const int* in_sizes, int n_in,
                              void* d_out, int out_size, void* d_ws, size_t ws_size,
                              hipStream_t stream) {
    const float*     x   = (const float*)d_in[0];
    const long long* ei  = (const long long*)d_in[1];   // int64 [2, E]
    const float*     ew  = (const float*)d_in[2];
    const float*     W1  = (const float*)d_in[3];
    const float*     b1  = (const float*)d_in[4];
    const float*     W2  = (const float*)d_in[5];
    const float*     b2  = (const float*)d_in[6];
    const float*     Ws  = (const float*)d_in[7];
    const float*     bs  = (const float*)d_in[8];
    const float*     Wl  = (const float*)d_in[9];
    const float*     bl  = (const float*)d_in[10];
    const float*     Wl2 = (const float*)d_in[11];
    const float*     bl2 = (const float*)d_in[12];

    const int  N  = in_sizes[0] / 128;     // 50000
    const long E  = in_sizes[1] / 2;       // 800000
    const long NF = (long)N * 128;
    const long NH = (long)N * 64;

    const long long* src = ei;
    const long long* dst = ei + E;

    float* out_final = (float*)d_out;            // [N,128]
    float* ls        = (float*)d_out + NF;       // [N,1]

    // Workspace layout (floats)
    float* buf0 = (float*)d_ws;          // [N,128] GEMM outputs (h1, later h2)
    float* buf1 = buf0 + NF;             // [N,128] agg1 -> h_neighbors -> h_new
    float* buf2 = buf1 + NF;             // [N,128] h_self -> agg2
    float* buf3 = buf2 + NF;             // [N,64]  hl
    float* buf4 = buf3 + NH;             // [N,64]  aggL -> ls_h
    float* dinv = buf4 + NH;             // [N]     deg -> dinv

    const int T = 256;   // 8 waves/block

    // 1) init: deg=1 (self loop), agg buffers = 0
    fill_kernel<<<blocksFor(N, T), T, 0, stream>>>(dinv, 1.0f, N);
    fill_kernel<<<4096, T, 0, stream>>>(buf1, 0.0f, NF);
    fill_kernel<<<4096, T, 0, stream>>>(buf4, 0.0f, NH);

    // 2) degrees, then dinv = rsqrt(deg)
    deg_kernel<<<blocksFor(E, T), T, 0, stream>>>(dst, ew, dinv, E);
    dinv_kernel<<<blocksFor(N, T), T, 0, stream>>>(dinv, N);

    // 3) dense projections (WMMA): h1 = x@W1, hl = x@Wl, h_self = x@Ws + bs
    const int tilesM  = (N + 15) / 16;           // 3125
    const int blk128  = tilesM;                  // 1 M-tile per block
    const int blk64   = (tilesM + 1) / 2;        // 2 M-tiles per block
    gemm_wmma_kernel<128><<<blk128, T, 0, stream>>>(x, W1, nullptr, buf0, N);
    gemm_wmma_kernel<64> <<<blk64,  T, 0, stream>>>(x, Wl, nullptr, buf3, N);
    gemm_wmma_kernel<128><<<blk128, T, 0, stream>>>(x, Ws, bs,      buf2, N);

    // 4) edge aggregation for conv1 and leader conv
    const int SCAT_BLOCKS = 8192;   // 65536 waves, grid-stride over E
    scatter_kernel<128><<<SCAT_BLOCKS, T, 0, stream>>>(buf0, buf1, src, dst, ew, dinv, E);
    scatter_kernel<64> <<<SCAT_BLOCKS, T, 0, stream>>>(buf3, buf4, src, dst, ew, dinv, E);

    // 5) epilogues: h_neighbors = relu(agg1 + dinv^2*h1 + b1); ls_h likewise
    epilogue_kernel<<<blocksFor(NF, T), T, 0, stream>>>(buf1, buf0, b1, dinv, buf1, NF, 128);
    epilogue_kernel<<<blocksFor(NH, T), T, 0, stream>>>(buf4, buf3, bl, dinv, buf4, NH, 64);

    // 6) leader score = sigmoid(ls_h @ Wl2 + bl2)  -> tail of d_out
    leader_kernel<<<blocksFor((long)N * WAVE, T), T, 0, stream>>>(buf4, Wl2, bl2, ls, N);

    // 7) gated fusion -> h_new (in place in buf1)
    fusion_kernel<<<blocksFor(NF, T), T, 0, stream>>>(buf1, buf2, ls, NF);

    // 8) conv2: h2 = h_new @ W2 (WMMA), aggregate, epilogue -> d_out
    gemm_wmma_kernel<128><<<blk128, T, 0, stream>>>(buf1, W2, nullptr, buf0, N);
    fill_kernel<<<4096, T, 0, stream>>>(buf2, 0.0f, NF);   // reuse h_self buffer as agg2
    scatter_kernel<128><<<SCAT_BLOCKS, T, 0, stream>>>(buf0, buf2, src, dst, ew, dinv, E);
    epilogue_kernel<<<blocksFor(NF, T), T, 0, stream>>>(buf2, buf0, b2, dinv, out_final, NF, 128);
}
